// GraphDqnModel_80556406604478
// MI455X (gfx1250) — matrix operand order repaired and checked by reading, not compiled, not run on hardware
//
#include <hip/hip_runtime.h>

// ---------------------------------------------------------------------------
// GraphDQN forward for MI455X (gfx1250, wave32).
// All contractions: v_wmma_f32_16x16x32_bf16, f32 accumulate.
// - One-time pack pass -> padded bf16 operands in consumer-matched layouts:
//   every hot fragment load is an unconditional per-lane-contiguous b128.
// - Conv stages activations into LDS via global_load_async_to_lds_b128
//   (ASYNCcnt) once per block; 4 co-tiles x 20 K-steps run from LDS.
// - Workload is HBM-bound (~0.8 GB moved => ~40us at 23.3 TB/s); bf16 staging
//   halves the dominant streams (objmap, fc_w, activations).
// B=64, E=4, N=100 (pad 112/128), P=4096, F=64, NACT=6.
// ---------------------------------------------------------------------------

typedef __attribute__((ext_vector_type(16))) __bf16 v16bf;
typedef __attribute__((ext_vector_type(8)))  __bf16 v8bf;
typedef __attribute__((ext_vector_type(8)))  float  v8f;

#define DEV __device__ __forceinline__

DEV __bf16 bf_zero() { return __builtin_bit_cast(__bf16, (unsigned short)0); }

DEV __bf16 f2bf(float f) {
    unsigned u = __builtin_bit_cast(unsigned, f);
    unsigned r = (u + 0x7FFFu + ((u >> 16) & 1u)) >> 16;   // RNE
    return __builtin_bit_cast(__bf16, (unsigned short)r);
}

DEV v8f wmma_bf16(v16bf a, v16bf b, v8f c) {
    return __builtin_amdgcn_wmma_f32_16x16x32_bf16(
        false, a, false, b, (short)0, c, false, false);
}

DEV void wait_asynccnt0() {
#if __has_builtin(__builtin_amdgcn_s_wait_asynccnt)
    __builtin_amdgcn_s_wait_asynccnt(0);
#else
    asm volatile("s_wait_asynccnt 0x0" ::: "memory");
#endif
}

// ---- fragment loaders (wave32; contiguous K = 16h+s in A and B, not the ISA table order) -----------------
// A 16x32 (contiguous K, as used here): lanes 0-15 row M=lane hold K=0..15; lanes 16-31 row M=lane-16
// hold K=16..31.  Contiguous K => two 16B loads per lane.
DEV v16bf load_A_ctg(const __bf16* p, int row0, int k0, long rs) {
    int lane = threadIdx.x & 31;
    int r    = row0 + (lane & 15);
    int hi   = lane >> 4;
    const __bf16* base = p + (long)r * rs + k0 + hi * 16;
    v8bf lo = *(const v8bf*)base;
    v8bf up = *(const v8bf*)(base + 8);
    return __builtin_shufflevector(lo, up, 0,1,2,3,4,5,6,7,8,9,10,11,12,13,14,15);
}

DEV v16bf load_A_ctg_f32(const float* p, int row0, int k0, long rs) {
    int lane = threadIdx.x & 31;
    int r    = row0 + (lane & 15);
    int hi   = lane >> 4;
    const float* base = p + (long)r * rs + k0 + hi * 16;
    v16bf a;
#pragma unroll
    for (int i = 0; i < 8; ++i) a[i] = f2bf(base[i]);
#pragma unroll
    for (int i = 0; i < 8; ++i) a[8 + i] = f2bf(base[8 + i]);
    return a;
}

// B 32x16 (same contiguous K): lanes 0-15 col N=lane hold K=0..15; lanes 16-31 hold K=16..31.
// Source in [col][k] (k contiguous, row stride cs) => one 32B load per lane.
DEV v16bf load_B_ctg(const __bf16* p, int k0, int col0, long cs) {
    int lane = threadIdx.x & 31;
    int c    = col0 + (lane & 15);
    int hi   = lane >> 4;
    return *(const v16bf*)(p + (long)c * cs + k0 + hi * 16);
}

// Source in [k][col] (k strided) => 16 unconditional u16 loads (coalesced
// across the 16 lanes of each half-wave).
DEV v16bf load_B_strided(const __bf16* p, int k0, int col0, long ks) {
    int lane = threadIdx.x & 31;
    int c    = col0 + (lane & 15);
    int hi   = lane >> 4;
    const __bf16* base = p + (long)(k0 + hi * 16) * ks + c;
    v16bf b;
#pragma unroll
    for (int i = 0; i < 16; ++i) b[i] = base[(long)i * ks];
    return b;
}

DEV v16bf load_B_strided_f32(const float* p, int k0, int col0, long ks) {
    int lane = threadIdx.x & 31;
    int c    = col0 + (lane & 15);
    int hi   = lane >> 4;
    const float* base = p + (long)(k0 + hi * 16) * ks + c;
    v16bf b;
#pragma unroll
    for (int i = 0; i < 16; ++i) b[i] = f2bf(base[(long)i * ks]);
    return b;
}

// Conv patch fragment out of the LDS-staged rows: tile[rr][x][ci], 64 ci
// contiguous per pixel. Border via clamped address + cndmask (no branching).
DEV v16bf load_patch_lds(const __bf16* tile, int rr, int x0, int dx, int ci0) {
    int lane = threadIdx.x & 31;
    int x = x0 + (lane & 15) + dx;
    bool ok = (unsigned)x < 64u;
    int xx = ok ? x : 0;
    v16bf v = *(const v16bf*)(tile + ((rr * 64 + xx) << 6) + ci0 + (lane >> 4) * 16);
#pragma unroll
    for (int i = 0; i < 16; ++i) v[i] = ok ? v[i] : bf_zero();
    return v;
}

// ---------------------------------------------------------------------------
// pack / init kernels (one pass each, negligible vs. main traffic)
__global__ void k_zero32(unsigned* __restrict__ p, long n) {
    long i = (long)blockIdx.x * blockDim.x + threadIdx.x;
    if (i < n) p[i] = 0u;
}
__global__ void k_pack_adj(const float* __restrict__ a, __bf16* __restrict__ o) {
    long i = (long)blockIdx.x * blockDim.x + threadIdx.x;
    if (i >= (long)256 * 112 * 128) return;
    long be = i / (112 * 128); int r = (int)((i / 128) % 112); int k = (int)(i & 127);
    o[i] = (r < 100 && k < 100) ? f2bf(a[be * 10000 + r * 100 + k]) : bf_zero();
}
__global__ void k_pack_w1t(const float* __restrict__ w, __bf16* __restrict__ o) {
    int i = blockIdx.x * blockDim.x + threadIdx.x;
    if (i >= 4 * 64 * 128) return;
    int e = i >> 13, c = (i >> 7) & 63, m = i & 127;
    o[i] = (m < 100) ? f2bf(w[e * 6400 + m * 64 + c]) : bf_zero();
}
__global__ void k_pack_obj(const float* __restrict__ a, __bf16* __restrict__ o) {
    long i = (long)blockIdx.x * blockDim.x + threadIdx.x;
    if (i >= (long)64 * 128 * 4096) return;
    long b = i >> 19; int n = (int)((i >> 12) & 127); int p = (int)(i & 4095);
    o[i] = (n < 100) ? f2bf(a[(b * 100 + n) * 4096 + p]) : bf_zero();
}
__global__ void k_pack_convw(const float* __restrict__ k3,
                             const float* __restrict__ k1,
                             __bf16* __restrict__ o) {
    int i = blockIdx.x * blockDim.x + threadIdx.x;
    if (i >= 64 * 640) return;
    int co = i / 640, k = i % 640, tap = k >> 6, ci = k & 63;
    float v = (tap < 9) ? k3[co * 576 + ci * 9 + tap] : k1[co * 64 + ci];
    o[i] = f2bf(v);
}
__global__ void k_pack_fcwt(const float* __restrict__ w, __bf16* __restrict__ o) {
    long i = (long)blockIdx.x * blockDim.x + threadIdx.x;
    if (i >= (long)64 * 262144) return;
    long j = i >> 18; long k = i & 262143;
    o[i] = (j < 50) ? f2bf(w[k * 50 + j]) : bf_zero();
}
__global__ void k_nocc(const float* __restrict__ objmap, float* __restrict__ nocc) {
    __shared__ float red[128];
    const float* p = objmap + (long)blockIdx.x * 4096;
    float s = 0.f;
    for (int i = threadIdx.x; i < 4096; i += 128) s += p[i];
    red[threadIdx.x] = s;
    __syncthreads();
    for (int off = 64; off > 0; off >>= 1) {
        if ((int)threadIdx.x < off) red[threadIdx.x] += red[threadIdx.x + off];
        __syncthreads();
    }
    if (threadIdx.x == 0) nocc[blockIdx.x] = red[0];
}

// ---------------------------------------------------------------------------
// kg[b,n,o] = relu(bias + 0.25 * sum_e adjp[b,e] @ St[b,e]^T), dual-layout out.
__global__ void k_gemm_adj(const __bf16* __restrict__ adjp,
                           const __bf16* __restrict__ St, long s_bstride,
                           const float* __restrict__ bias,
                           __bf16* __restrict__ kg,      // [b][n(112)][f]
                           __bf16* __restrict__ kgT) {   // [b][f][n(128)]
    int row0 = blockIdx.x * 16, col0 = blockIdx.y * 16, b = blockIdx.z;
    v8f acc = {};
    for (int e = 0; e < 4; ++e) {
        const __bf16* ap = adjp + ((long)b * 4 + e) * 112 * 128;
        const __bf16* bp = St + s_bstride * b + (long)e * 64 * 128;
#pragma unroll
        for (int kt = 0; kt < 4; ++kt) {
            acc = wmma_bf16(load_A_ctg(ap, row0, kt * 32, 128),
                            load_B_ctg(bp, kt * 32, col0, 128), acc);
        }
    }
    int lane = threadIdx.x & 31;
    int c = col0 + (lane & 15), hi = lane >> 4;
#pragma unroll
    for (int r = 0; r < 8; ++r) {
        int row = row0 + r + hi * 8;
        float v = 0.25f * acc[r] + bias[c];
        __bf16 h = (row < 100 && v > 0.f) ? f2bf(v) : bf_zero();
        kg[((long)b * 112 + row) * 64 + c] = h;
        kgT[((long)b * 64 + c) * 128 + row] = h;
    }
}

// St[b,e][o][m] = sum_i kg[b,m,i] * w[e,i,o]
__global__ void k_support(const __bf16* __restrict__ kg,
                          const float* __restrict__ w,
                          __bf16* __restrict__ St) {
    int row0 = blockIdx.x * 16, col0 = blockIdx.y * 16;
    int b = blockIdx.z >> 2, e = blockIdx.z & 3;
    const __bf16* ap = kg + (long)b * 112 * 64;
    const float*  bp = w + (long)e * 64 * 64;
    v8f acc = {};
#pragma unroll
    for (int kt = 0; kt < 2; ++kt) {
        acc = wmma_bf16(load_A_ctg(ap, row0, kt * 32, 64),
                        load_B_strided_f32(bp, kt * 32, col0, 64), acc);
    }
    int lane = threadIdx.x & 31;
    int c = col0 + (lane & 15), hi = lane >> 4;
#pragma unroll
    for (int r = 0; r < 8; ++r) {
        int row = row0 + r + hi * 8;  // = m
        St[(((long)b * 4 + e) * 64 + c) * 128 + row] =
            (row < 100) ? f2bf(acc[r]) : bf_zero();
    }
}

// state_t[b][p][f] = sum_n kgT[b][f][n] * objp[b][n][p]
__global__ void k_broadcast(const __bf16* __restrict__ kgT,
                            const __bf16* __restrict__ objp,
                            __bf16* __restrict__ outT) {
    int wv = threadIdx.x >> 5;
    int pt = blockIdx.x * 4 + wv;
    int row0 = blockIdx.y * 16, col0 = pt * 16, b = blockIdx.z;
    const __bf16* ap = kgT + (long)b * 64 * 128;
    const __bf16* bp = objp + (long)b * 128 * 4096;
    v8f acc = {};
#pragma unroll
    for (int kt = 0; kt < 4; ++kt) {
        acc = wmma_bf16(load_A_ctg(ap, row0, kt * 32, 128),
                        load_B_strided(bp, kt * 32, col0, 4096), acc);
    }
    int lane = threadIdx.x & 31;
    int c = col0 + (lane & 15), hi = lane >> 4;
#pragma unroll
    for (int r = 0; r < 8; ++r) {
        int row = row0 + r + hi * 8;  // = f
        outT[((long)b * 4096 + c) * 64 + row] = f2bf(acc[r]);
    }
}

// Fused conv: out[b,co,p] = relu(b3+b1 + conv3x3(xa) + conv1x1(xb)).
// One block = one image row (64 px) x all 64 co. Activation rows h-1,h,h+1
// (xa) and h (xb) staged once into LDS via async-to-LDS; all 4 co-tiles and
// all taps consume from LDS (40x reuse). grid (64 rows, B), block 128.
__global__ void k_conv(const __bf16* __restrict__ xat,
                       const __bf16* __restrict__ xbt,
                       const __bf16* __restrict__ Wp,
                       const float* __restrict__ b3,
                       const float* __restrict__ b1,
                       __bf16* __restrict__ out,
                       __bf16* __restrict__ outT) {
    __shared__ __align__(64) __bf16 tile[4 * 64 * 64];   // 32 KB
    int h = blockIdx.x, b = blockIdx.y;
    int tid = threadIdx.x;

    // ---- stage: rows {h-1,h,h+1} of xat and row h of xbt (each 8 KB linear)
#pragma unroll
    for (int i = 0; i < 16; ++i) {
        int c = i * 128 + tid;            // 16-byte chunk id, 2048 total
        int r = c >> 9, q = c & 511;      // tile row 0..3, chunk within row
        int hr = h + r - 1;
        bool is_b = (r == 3);
        bool valid = is_b || ((unsigned)hr < 64u);
        __bf16* dst = &tile[c << 3];
        if (valid) {
            const __bf16* src = is_b
                ? (xbt + ((long)b * 4096 + h * 64) * 64 + (q << 3))
                : (xat + ((long)b * 4096 + hr * 64) * 64 + (q << 3));
            unsigned lds = (unsigned)(unsigned long long)(const void*)dst;
            unsigned long long ga = (unsigned long long)(const void*)src;
            asm volatile("global_load_async_to_lds_b128 %0, %1, off"
                         :: "v"(lds), "v"(ga) : "memory");
        } else {
            v8bf z = {};
            *(v8bf*)dst = z;               // zero halo outside the image
        }
    }
    wait_asynccnt0();
    __syncthreads();

    // ---- compute: wave -> 16-px quarter; 20 K-steps x 4 co-tiles
    int wv = tid >> 5, lane = tid & 31;
    int x0 = wv * 16;
    v8f acc0 = {}, acc1 = {}, acc2 = {}, acc3 = {};
#pragma unroll
    for (int kt = 0; kt < 20; ++kt) {
        int tap = kt >> 1, ci0 = (kt & 1) * 32;
        int rr = (tap < 9) ? (tap / 3) : 3;        // LDS row (dy+1, or 1x1 row)
        int dx = (tap < 9) ? (tap % 3 - 1) : 0;
        v16bf bb = load_patch_lds(tile, rr, x0, dx, ci0);
        acc0 = wmma_bf16(load_A_ctg(Wp,  0, kt * 32, 640), bb, acc0);
        acc1 = wmma_bf16(load_A_ctg(Wp, 16, kt * 32, 640), bb, acc1);
        acc2 = wmma_bf16(load_A_ctg(Wp, 32, kt * 32, 640), bb, acc2);
        acc3 = wmma_bf16(load_A_ctg(Wp, 48, kt * 32, 640), bb, acc3);
    }
    int c = h * 64 + x0 + (lane & 15);    // global pixel index
    int hi = lane >> 4;
    v8f accs[4] = {acc0, acc1, acc2, acc3};
#pragma unroll
    for (int ct = 0; ct < 4; ++ct) {
#pragma unroll
        for (int r = 0; r < 8; ++r) {
            int row = ct * 16 + r + hi * 8;       // = co
            float v = accs[ct][r] + b3[row] + b1[row];
            __bf16 hv = v > 0.f ? f2bf(v) : bf_zero();
            out[((long)b * 64 + row) * 4096 + c] = hv;
            if (outT) outT[((long)b * 4096 + c) * 64 + row] = hv;
        }
    }
}

// pooled[b,n,f] = sum_p objp[b,n,p] * skip[b,f,p]; 4-wave split-K, LDS reduce
__global__ void k_pool(const __bf16* __restrict__ objp,
                       const __bf16* __restrict__ skip,
                       float* __restrict__ pooled) {
    __shared__ float red[4][8][32];
    int wv = threadIdx.x >> 5, lane = threadIdx.x & 31;
    int row0 = blockIdx.x * 16, col0 = blockIdx.y * 16, b = blockIdx.z;
    const __bf16* ap = objp + (long)b * 128 * 4096;
    const __bf16* bp = skip + (long)b * 64 * 4096;
    v8f acc = {};
    for (int kt = wv; kt < 128; kt += 4) {
        acc = wmma_bf16(load_A_ctg(ap, row0, kt * 32, 4096),
                        load_B_ctg(bp, kt * 32, col0, 4096), acc);
    }
#pragma unroll
    for (int r = 0; r < 8; ++r) red[wv][r][lane] = acc[r];
    __syncthreads();
    if (wv == 0) {
#pragma unroll
        for (int r = 0; r < 8; ++r) {
            float v = red[0][r][lane] + red[1][r][lane] +
                      red[2][r][lane] + red[3][r][lane];
            int row = row0 + r + (lane >> 4) * 8;
            int c   = col0 + (lane & 15);
            if (row < 100) pooled[((long)b * 112 + row) * 64 + c] = v;
        }
    }
}

// kg[b,n,f] = (pooled[b] @ pool_w)[n,f] / nocc[b,n], dual-layout out
__global__ void k_pool_proj(const float* __restrict__ pooled,
                            const float* __restrict__ pw,
                            const float* __restrict__ nocc,
                            __bf16* __restrict__ kg,
                            __bf16* __restrict__ kgT) {
    int row0 = blockIdx.x * 16, col0 = blockIdx.y * 16, b = blockIdx.z;
    const float* ap = pooled + (long)b * 112 * 64;
    v8f acc = {};
#pragma unroll
    for (int kt = 0; kt < 2; ++kt) {
        acc = wmma_bf16(load_A_ctg_f32(ap, row0, kt * 32, 64),
                        load_B_strided_f32(pw, kt * 32, col0, 64), acc);
    }
    int lane = threadIdx.x & 31;
    int c = col0 + (lane & 15), hi = lane >> 4;
#pragma unroll
    for (int r = 0; r < 8; ++r) {
        int row = row0 + r + hi * 8;
        int ri = row < 100 ? row : 0;
        float v = acc[r] / nocc[b * 100 + ri];
        __bf16 h = (row < 100) ? f2bf(v) : bf_zero();
        kg[((long)b * 112 + row) * 64 + c] = h;
        kgT[((long)b * 64 + c) * 128 + row] = h;
    }
}

// Split-K fc: part[ks][b][j] — fully vectorized, prefetch on both streams.
__global__ void k_fc(const __bf16* __restrict__ flat,
                     const __bf16* __restrict__ fcwt,
                     float* __restrict__ part) {
    int ks = blockIdx.x, col0 = blockIdx.y * 16, row0 = blockIdx.z * 16;
    long kbase = (long)ks * 4096;
    int lane = threadIdx.x & 31;
    int hi = lane >> 4;
    int c = col0 + (lane & 15);
    int rr = row0 + (lane & 15);
    const __bf16* aq = flat + (long)rr * 262144 + kbase + hi * 16;
    const __bf16* bq = fcwt + (long)c * 262144 + kbase + hi * 16;
    v8f acc = {};
    for (int k0 = 0; k0 < 4096; k0 += 32) {
        if (k0 + 2048 < 4096) {
            __builtin_prefetch(bq + k0 + 2048, 0, 1);
            __builtin_prefetch(aq + k0 + 2048, 0, 1);
        }
        v8bf lo = *(const v8bf*)(aq + k0);
        v8bf up = *(const v8bf*)(aq + k0 + 8);
        v16bf a = __builtin_shufflevector(lo, up,
                      0,1,2,3,4,5,6,7,8,9,10,11,12,13,14,15);
        v16bf bb = *(const v16bf*)(bq + k0);
        acc = wmma_bf16(a, bb, acc);
    }
#pragma unroll
    for (int r = 0; r < 8; ++r) {
        int row = row0 + r + hi * 8;
        part[((long)ks * 64 + row) * 64 + c] = acc[r];
    }
}

__global__ void k_hidden(const float* __restrict__ part,
                         const float* __restrict__ fcb,
                         float* __restrict__ hid) {
    int idx = blockIdx.x * blockDim.x + threadIdx.x;
    if (idx >= 64 * 64) return;
    int j = idx & 63;
    float s = 0.f;
    for (int ks = 0; ks < 64; ++ks) s += part[(long)ks * 4096 + idx];
    float v = (j < 50) ? s + fcb[j] : 0.f;
    hid[idx] = v > 0.f ? v : 0.f;
}

__global__ void k_head(const float* __restrict__ hid,
                       const float* __restrict__ hw,
                       const float* __restrict__ hb,
                       float* __restrict__ out) {
    int t = threadIdx.x;
    if (t >= 384) return;
    int b = t / 6, a = t % 6;
    float s = hb[a];
    for (int j = 0; j < 50; ++j) s += hid[b * 64 + j] * hw[j * 6 + a];
    out[t] = s;
}

// ---------------------------------------------------------------------------
extern "C" void kernel_launch(void* const* d_in, const int* in_sizes, int n_in,
                              void* d_out, int out_size, void* d_ws,
                              size_t ws_size, hipStream_t stream) {
    (void)in_sizes; (void)n_in; (void)out_size; (void)ws_size;
    const float* adj    = (const float*)d_in[0];
    const float* objmap = (const float*)d_in[1];
    const float* w1  = (const float*)d_in[2];  const float* bg1 = (const float*)d_in[3];
    const float* w2  = (const float*)d_in[4];  const float* bg2 = (const float*)d_in[5];
    const float* w3  = (const float*)d_in[6];  const float* bg3 = (const float*)d_in[7];
    const float* w4  = (const float*)d_in[8];  const float* bg4 = (const float*)d_in[9];
    const float* ck1 = (const float*)d_in[10]; const float* cb1 = (const float*)d_in[11];
    const float* ck2 = (const float*)d_in[12]; const float* cb2 = (const float*)d_in[13];
    const float* ck3 = (const float*)d_in[14]; const float* cb3 = (const float*)d_in[15];
    const float* ck4 = (const float*)d_in[16]; const float* cb4 = (const float*)d_in[17];
    const float* pool_w = (const float*)d_in[18];
    const float* fc_w = (const float*)d_in[19]; const float* fc_b = (const float*)d_in[20];
    const float* head_w = (const float*)d_in[21]; const float* head_b = (const float*)d_in[22];
    float* out = (float*)d_out;

    char* ws = (char*)d_ws;
    size_t off = 0;
    auto carve = [&](size_t bytes) -> void* {
        void* p = ws + off;
        off += (bytes + 255) & ~(size_t)255;
        return p;
    };
    size_t z0 = 0;
    __bf16* kg1  = (__bf16*)carve((size_t)64 * 112 * 64 * 2);
    __bf16* kg1T = (__bf16*)carve((size_t)64 * 64 * 128 * 2);
    __bf16* kg2  = (__bf16*)carve((size_t)64 * 112 * 64 * 2);
    __bf16* kg2T = (__bf16*)carve((size_t)64 * 64 * 128 * 2);
    __bf16* St   = (__bf16*)carve((size_t)64 * 4 * 64 * 128 * 2);
    float*  pooled = (float*)carve((size_t)64 * 112 * 64 * 4);
    size_t z1 = off;
    __bf16* adjp = (__bf16*)carve((size_t)64 * 4 * 112 * 128 * 2);
    __bf16* w1t  = (__bf16*)carve((size_t)4 * 64 * 128 * 2);
    __bf16* Wp1  = (__bf16*)carve((size_t)64 * 640 * 2);
    __bf16* Wp2  = (__bf16*)carve((size_t)64 * 640 * 2);
    __bf16* objp = (__bf16*)carve((size_t)64 * 128 * 4096 * 2);
    __bf16* fcwt = (__bf16*)carve((size_t)64 * 262144 * 2);
    __bf16* big1t = (__bf16*)carve((size_t)64 * 4096 * 64 * 2); // state_t/state2_t
    __bf16* big2  = (__bf16*)carve((size_t)64 * 64 * 4096 * 2); // skip [f][p]
    __bf16* big2t = (__bf16*)carve((size_t)64 * 4096 * 64 * 2); // skip_t [p][f]
    __bf16* big3  = (__bf16*)carve((size_t)64 * 64 * 4096 * 2); // final [f][p]
    float* nocc = (float*)carve((size_t)64 * 100 * 4);
    float* part = (float*)carve((size_t)64 * 64 * 64 * 4);
    float* hid  = (float*)carve((size_t)64 * 64 * 4);

    dim3 b32(32), b128(128), b256(256);
    dim3 g_nf(7, 4, 64), g_pf(64, 4, 64), g_cv(64, 64);

    long zn = (long)(z1 - z0) / 4;
    k_zero32<<<dim3((unsigned)((zn + 255) / 256)), b256, 0, stream>>>((unsigned*)(ws + z0), zn);
    k_pack_adj<<<dim3((256 * 112 * 128) / 256), b256, 0, stream>>>(adj, adjp);
    k_pack_w1t<<<dim3(128), b256, 0, stream>>>(w1, w1t);
    k_pack_convw<<<dim3(160), b256, 0, stream>>>(ck1, ck2, Wp1);
    k_pack_convw<<<dim3(160), b256, 0, stream>>>(ck3, ck4, Wp2);
    k_pack_obj<<<dim3((64 * 128 * 4096) / 256), b256, 0, stream>>>(objmap, objp);
    k_pack_fcwt<<<dim3((64 * 262144) / 256), b256, 0, stream>>>(fc_w, fcwt);
    k_nocc<<<dim3(64 * 100), b128, 0, stream>>>(objmap, nocc);

    const long sstride = (long)4 * 64 * 128;
    k_gemm_adj<<<g_nf, b32, 0, stream>>>(adjp, w1t, 0L, bg1, kg1, kg1T);
    k_support<<<dim3(7, 4, 256), b32, 0, stream>>>(kg1, w2, St);
    k_gemm_adj<<<g_nf, b32, 0, stream>>>(adjp, St, sstride, bg2, kg2, kg2T);
    k_broadcast<<<g_pf, b128, 0, stream>>>(kg2T, objp, big1t);
    k_conv<<<g_cv, b128, 0, stream>>>(big1t, big1t, Wp1, cb1, cb2, big2, big2t);
    k_pool<<<g_nf, b128, 0, stream>>>(objp, big2, pooled);
    k_pool_proj<<<g_nf, b32, 0, stream>>>(pooled, pool_w, nocc, kg1, kg1T);
    k_support<<<dim3(7, 4, 256), b32, 0, stream>>>(kg1, w3, St);
    k_gemm_adj<<<g_nf, b32, 0, stream>>>(adjp, St, sstride, bg3, kg2, kg2T);
    k_support<<<dim3(7, 4, 256), b32, 0, stream>>>(kg2, w4, St);
    k_gemm_adj<<<g_nf, b32, 0, stream>>>(adjp, St, sstride, bg4, kg1, kg1T);
    k_broadcast<<<g_pf, b128, 0, stream>>>(kg1T, objp, big1t);
    k_conv<<<g_cv, b128, 0, stream>>>(big2t, big1t, Wp2, cb3, cb4, big3, nullptr);
    k_fc<<<dim3(64, 4, 4), b32, 0, stream>>>(big3, fcwt, part);
    k_hidden<<<dim3(16), b256, 0, stream>>>(part, fc_b, hid);
    k_head<<<dim3(1), dim3(384), 0, stream>>>(hid, head_w, head_b, out);
}
